// GroupedQueryAttention_83717502534082
// MI455X (gfx1250) — matrix operand order repaired and checked by reading, not compiled, not run on hardware
//
#include <hip/hip_runtime.h>

typedef __bf16 bf16;
typedef __attribute__((ext_vector_type(16))) __bf16 v16bf;
typedef __attribute__((ext_vector_type(8)))  __bf16 v8bf;
typedef __attribute__((ext_vector_type(4)))  __bf16 v4bf;
typedef __attribute__((ext_vector_type(8)))  float  v8f;

#define D_MODEL 2048
#define KV_DIM  512
#define SEQ     1024
#define BATCH   2
#define NHEADS  32
#define NKVH    8
#define HDIM    64

__device__ __forceinline__ bf16 f2bf(float f) {
  unsigned u = __builtin_bit_cast(unsigned, f);
  u = (u + 0x7FFFu + ((u >> 16) & 1u)) >> 16;
  unsigned short s = (unsigned short)u;
  return __builtin_bit_cast(bf16, s);
}

__device__ __forceinline__ v16bf cat16(v8bf lo, v8bf hi) {
  return __builtin_shufflevector(lo, hi, 0,1,2,3,4,5,6,7,8,9,10,11,12,13,14,15);
}

// low 32 bits of a flat pointer to __shared__ = LDS byte offset (shared->flat
// addrspacecast is {SHARED_BASE, offset})
__device__ __forceinline__ unsigned lds_off(const void* p) {
  return (unsigned)(uintptr_t)p;
}

// async direct-to-LDS 16B copy: vdst = LDS byte offset, vaddr = 32-bit byte
// offset added to 64-bit SGPR base. Tracked by ASYNCcnt.
__device__ __forceinline__ void async_b128(unsigned ldsaddr, unsigned voff, const void* sbase) {
  asm volatile("global_load_async_to_lds_b128 %0, %1, %2"
               :: "v"(ldsaddr), "v"(voff), "s"(sbase) : "memory");
}

// ---------------- fp32 -> bf16 cast ----------------
__global__ void cast_f32_bf16(const float* __restrict__ in, bf16* __restrict__ out, int n) {
  int i = blockIdx.x * blockDim.x + threadIdx.x;
  if (i < n) out[i] = f2bf(in[i]);
}

// ---------------- V transpose: Vp[B*S, KV_DIM] -> Vt[B, NKVH, HDIM, SEQ] ----------------
__global__ void transpose_v(const bf16* __restrict__ vp, bf16* __restrict__ vt) {
  int i = blockIdx.x * blockDim.x + threadIdx.x;   // B*NKVH*HDIM*SEQ = 2^20
  int n   = i & (SEQ - 1);
  int d   = (i >> 10) & (HDIM - 1);
  int kvh = (i >> 16) & (NKVH - 1);
  int b   = i >> 19;
  vt[i] = vp[((size_t)(b * SEQ + n)) * KV_DIM + kvh * HDIM + d];
}

// ---------------- bf16 GEMM: C[M,N] = A[M,K] @ W[N,K]^T + bias ----------------
// BM=128, BN=64, BK=32; 256 threads = 8 waves in 4(M) x 2(N), 32x32 per wave.
// Tiles stream into double-buffered LDS via ASYNC direct-to-LDS loads.
template<bool OUT_BF16>
__global__ __launch_bounds__(256) void gemm_bf16(
    const bf16* __restrict__ A, const bf16* __restrict__ W,
    const float* __restrict__ bias, void* __restrict__ Cout,
    int M, int N, int K)
{
  __shared__ __align__(16) bf16 At[2][128 * 32];
  __shared__ __align__(16) bf16 Bt[2][64 * 32];

  const int m0 = blockIdx.y * 128;
  const int n0 = blockIdx.x * 64;
  const int tid = threadIdx.x;
  const int wave = tid >> 5, lane = tid & 31;
  const int ln = lane & 15, hi = lane >> 4;
  const int wm = (wave & 3) * 32;
  const int wn = (wave >> 2) * 32;
  const int kbase = hi * 8;

  // per-thread chunk coordinates (16B chunks)
  const int ar0 = tid >> 2,        akc0 = (tid & 3) * 8;          // A chunk 0
  const int ar1 = (tid + 256) >> 2, akc1 = ((tid + 256) & 3) * 8;  // A chunk 1
  const int br  = tid >> 2,        bkc  = (tid & 3) * 8;          // B chunk
  const bf16* Abase = A + (size_t)m0 * K;
  const bf16* Wbase = W + (size_t)n0 * K;
  const unsigned aoff0 = (unsigned)((ar0 * K + akc0) * 2);
  const unsigned aoff1 = (unsigned)((ar1 * K + akc1) * 2);
  const unsigned boff  = (unsigned)((br  * K + bkc ) * 2);

  auto issue_tile = [&](int k0, int buf) {
    unsigned kb = (unsigned)(k0 * 2);
    async_b128(lds_off(&At[buf][ar0 * 32 + akc0]), aoff0 + kb, Abase);
    async_b128(lds_off(&At[buf][ar1 * 32 + akc1]), aoff1 + kb, Abase);
    async_b128(lds_off(&Bt[buf][br  * 32 + bkc ]), boff  + kb, Wbase);
  };

  v8f acc[2][2] = {};

  issue_tile(0, 0);
  int cur = 0;
  for (int k0 = 0; k0 < K; k0 += 32) {
    if (k0 + 32 < K) {
      issue_tile(k0 + 32, cur ^ 1);
      asm volatile("s_wait_asynccnt 3" ::: "memory");   // tile k0 landed
    } else {
      asm volatile("s_wait_asynccnt 0" ::: "memory");
    }
    __syncthreads();

    #pragma unroll
    for (int ti = 0; ti < 2; ++ti) {
      int ar = wm + ti * 16 + ln;
      v8bf alo = *(const v8bf*)&At[cur][ar * 32 + kbase];
      v8bf ahi = *(const v8bf*)&At[cur][ar * 32 + kbase + 16];
      v16bf af = cat16(alo, ahi);
      #pragma unroll
      for (int tj = 0; tj < 2; ++tj) {
        int brr = wn + tj * 16 + ln;
        v8bf blo = *(const v8bf*)&Bt[cur][brr * 32 + kbase];
        v8bf bhi = *(const v8bf*)&Bt[cur][brr * 32 + kbase + 16];
        v16bf bfr = cat16(blo, bhi);
        acc[ti][tj] = __builtin_amdgcn_wmma_f32_16x16x32_bf16(
            false, af, false, bfr, (short)0, acc[ti][tj], false, false);
      }
    }
    __syncthreads();
    cur ^= 1;
  }

  #pragma unroll
  for (int ti = 0; ti < 2; ++ti)
    #pragma unroll
    for (int tj = 0; tj < 2; ++tj) {
      int col = n0 + wn + tj * 16 + ln;
      float bv = bias[col];
      #pragma unroll
      for (int i = 0; i < 8; ++i) {
        int r = m0 + wm + ti * 16 + i + 8 * hi;
        float v = acc[ti][tj][i] + bv;
        if (OUT_BF16) ((bf16*)Cout)[(size_t)r * N + col] = f2bf(v);
        else          ((float*)Cout)[(size_t)r * N + col] = v;
      }
    }
}

// ---------------- fused GQA attention ----------------
// One workgroup per (b, h, 16-row query tile). 8 waves, each owns 128 key cols.
__global__ __launch_bounds__(256) void attn_kernel(
    const bf16* __restrict__ Qp,   // [B, S, D_MODEL]
    const bf16* __restrict__ Kp,   // [B, S, KV_DIM]
    const bf16* __restrict__ Vt,   // [B, NKVH, HDIM, S]
    float* __restrict__ attn_out,  // [B, H, S, S]
    bf16* __restrict__ ctx)        // [B, S, D_MODEL]
{
  __shared__ __align__(16) bf16 attnLds[8 * 16 * 128];   // per-wave bf16 probs
  __shared__ float outacc[16 * 64];
  __shared__ float redmax[8][16];
  __shared__ float redsum[8][16];

  const int bid = blockIdx.x;
  const int qt = bid & 63;
  const int h  = (bid >> 6) & 31;
  const int b  = bid >> 11;
  const int m0 = qt * 16;

  const int tid = threadIdx.x;
  const int wave = tid >> 5, lane = tid & 31;
  const int ln = lane & 15, hi = lane >> 4;
  const int kbase = hi * 8;
  const int n0w = wave * 128;

  #pragma unroll
  for (int j = 0; j < 4; ++j) outacc[tid * 4 + j] = 0.f;

  const bf16* Qb = Qp + ((size_t)(b * SEQ + m0)) * D_MODEL + h * HDIM;
  const bf16* Kb = Kp + ((size_t)b * SEQ) * KV_DIM + (h >> 2) * HDIM;
  const bf16* Vb = Vt + ((size_t)((b * NKVH + (h >> 2)) * HDIM)) * SEQ;
  float* Ab = attn_out + ((size_t)((b * NHEADS + h) * SEQ + m0)) * SEQ;

  // Q fragments (16x32 each, two K-chunks of head_dim 64)
  v16bf qf[2];
  #pragma unroll
  for (int s2 = 0; s2 < 2; ++s2) {
    v8bf lo = *(const v8bf*)&Qb[(size_t)ln * D_MODEL + s2 * 32 + kbase];
    v8bf hh = *(const v8bf*)&Qb[(size_t)ln * D_MODEL + s2 * 32 + kbase + 16];
    qf[s2] = cat16(lo, hh);
  }

  // scores: 8 tiles of 16x16 per wave, fp32 accum in registers
  v8f sc[8];
  #pragma unroll
  for (int c = 0; c < 8; ++c) {
    v8f a = {};
    int n = n0w + c * 16 + ln;
    #pragma unroll
    for (int s2 = 0; s2 < 2; ++s2) {
      v8bf blo = *(const v8bf*)&Kb[(size_t)n * KV_DIM + s2 * 32 + kbase];
      v8bf bhi = *(const v8bf*)&Kb[(size_t)n * KV_DIM + s2 * 32 + kbase + 16];
      v16bf bfr = cat16(blo, bhi);
      a = __builtin_amdgcn_wmma_f32_16x16x32_bf16(false, qf[s2], false, bfr,
                                                  (short)0, a, false, false);
    }
    #pragma unroll
    for (int i = 0; i < 8; ++i) a[i] *= 0.125f;   // 1/sqrt(64)
    sc[c] = a;
  }

  // ---- softmax over 1024 cols (intra-wave shfl + cross-wave LDS) ----
  float rmax[8];
  #pragma unroll
  for (int i = 0; i < 8; ++i) {
    float m = sc[0][i];
    #pragma unroll
    for (int c = 1; c < 8; ++c) m = fmaxf(m, sc[c][i]);
    #pragma unroll
    for (int off = 1; off < 16; off <<= 1) m = fmaxf(m, __shfl_xor(m, off, 32));
    rmax[i] = m;
  }
  if (lane == 0) {
    for (int i = 0; i < 8; ++i) redmax[wave][i] = rmax[i];
  }
  if (lane == 16) {
    for (int i = 0; i < 8; ++i) redmax[wave][8 + i] = rmax[i];
  }
  __syncthreads();

  float gmax[8];
  #pragma unroll
  for (int i = 0; i < 8; ++i) {
    float m = redmax[0][i + 8 * hi];
    #pragma unroll
    for (int w = 1; w < 8; ++w) m = fmaxf(m, redmax[w][i + 8 * hi]);
    gmax[i] = m;
  }

  float rsum[8];
  #pragma unroll
  for (int i = 0; i < 8; ++i) rsum[i] = 0.f;
  #pragma unroll
  for (int c = 0; c < 8; ++c)
    #pragma unroll
    for (int i = 0; i < 8; ++i) {
      float e = __expf(sc[c][i] - gmax[i]);
      sc[c][i] = e;
      rsum[i] += e;
    }
  #pragma unroll
  for (int i = 0; i < 8; ++i) {
    #pragma unroll
    for (int off = 1; off < 16; off <<= 1) rsum[i] += __shfl_xor(rsum[i], off, 32);
  }
  if (lane == 0) {
    for (int i = 0; i < 8; ++i) redsum[wave][i] = rsum[i];
  }
  if (lane == 16) {
    for (int i = 0; i < 8; ++i) redsum[wave][8 + i] = rsum[i];
  }
  __syncthreads();

  float ginv[8];
  #pragma unroll
  for (int i = 0; i < 8; ++i) {
    float s = 0.f;
    #pragma unroll
    for (int w = 0; w < 8; ++w) s += redsum[w][i + 8 * hi];
    ginv[i] = __frcp_rn(s);
  }

  // write fp32 attn to d_out + bf16 probs to own LDS strip
  #pragma unroll
  for (int c = 0; c < 8; ++c)
    #pragma unroll
    for (int i = 0; i < 8; ++i) {
      int mrow = i + 8 * hi;
      float v = sc[c][i] * ginv[i];
      Ab[(size_t)mrow * SEQ + n0w + c * 16 + ln] = v;
      attnLds[(wave * 16 + mrow) * 128 + c * 16 + ln] = f2bf(v);
    }

  // ---- PV: out[16,64] partial per wave over its 128 keys ----
  v16bf pf[4];
  #pragma unroll
  for (int kt = 0; kt < 4; ++kt) {
    const bf16* base = &attnLds[(wave * 16 + ln) * 128 + kt * 32];
    v8bf lo = *(const v8bf*)&base[kbase];
    v8bf hh = *(const v8bf*)&base[kbase + 16];
    pf[kt] = cat16(lo, hh);
  }
  #pragma unroll
  for (int dt = 0; dt < 4; ++dt) {
    v8f a = {};
    int d = dt * 16 + ln;
    #pragma unroll
    for (int kt = 0; kt < 4; ++kt) {
      const bf16* vb = &Vb[(size_t)d * SEQ + n0w + kt * 32 + kbase];
      v8bf lo = *(const v8bf*)vb;
      v8bf hh = *(const v8bf*)(vb + 16);
      v16bf bfr = cat16(lo, hh);
      a = __builtin_amdgcn_wmma_f32_16x16x32_bf16(false, pf[kt], false, bfr,
                                                  (short)0, a, false, false);
    }
    #pragma unroll
    for (int i = 0; i < 8; ++i)
      atomicAdd(&outacc[(i + 8 * hi) * 64 + dt * 16 + ln], a[i]);  // ds_add_f32
  }
  __syncthreads();

  // write bf16 context tile
  bf16* cb = ctx + ((size_t)(b * SEQ + m0)) * D_MODEL + h * HDIM;
  {
    int idx = tid * 4;
    int m = idx >> 6, d = idx & 63;
    v4bf v;
    v[0] = f2bf(outacc[idx + 0]);
    v[1] = f2bf(outacc[idx + 1]);
    v[2] = f2bf(outacc[idx + 2]);
    v[3] = f2bf(outacc[idx + 3]);
    *(v4bf*)&cb[(size_t)m * D_MODEL + d] = v;
  }
}

// ---------------- launch ----------------
extern "C" void kernel_launch(void* const* d_in, const int* in_sizes, int n_in,
                              void* d_out, int out_size, void* d_ws, size_t ws_size,
                              hipStream_t stream) {
  const float* q  = (const float*)d_in[0];
  const float* k  = (const float*)d_in[1];
  const float* v  = (const float*)d_in[2];
  const float* Wq = (const float*)d_in[3];
  const float* bq = (const float*)d_in[4];
  const float* Wk = (const float*)d_in[5];
  const float* bk = (const float*)d_in[6];
  const float* Wv = (const float*)d_in[7];
  const float* bv = (const float*)d_in[8];
  const float* Wo = (const float*)d_in[9];
  const float* bo = (const float*)d_in[10];
  float* out = (float*)d_out;

  const size_t NM = (size_t)BATCH * SEQ;   // 2048 token rows
  char* ws = (char*)d_ws;
  size_t off = 0;
  auto alloc = [&](size_t bytes) -> void* {
    void* p = ws + off;
    off = (off + bytes + 255) & ~(size_t)255;
    return p;
  };
  bf16* qb  = (bf16*)alloc(NM * D_MODEL * 2);
  bf16* kb  = (bf16*)alloc(NM * D_MODEL * 2);
  bf16* vb  = (bf16*)alloc(NM * D_MODEL * 2);
  bf16* wqb = (bf16*)alloc((size_t)D_MODEL * D_MODEL * 2);
  bf16* wkb = (bf16*)alloc((size_t)KV_DIM  * D_MODEL * 2);
  bf16* wvb = (bf16*)alloc((size_t)KV_DIM  * D_MODEL * 2);
  bf16* wob = (bf16*)alloc((size_t)D_MODEL * D_MODEL * 2);
  bf16* Qp  = (bf16*)alloc(NM * D_MODEL * 2);
  bf16* Kp  = (bf16*)alloc(NM * KV_DIM  * 2);
  bf16* Vp  = (bf16*)alloc(NM * KV_DIM  * 2);
  bf16* Vt  = (bf16*)alloc(NM * KV_DIM  * 2);
  bf16* ctx = (bf16*)alloc(NM * D_MODEL * 2);

  auto cast = [&](const float* src, bf16* dst, size_t n) {
    cast_f32_bf16<<<dim3((unsigned)((n + 255) / 256)), 256, 0, stream>>>(src, dst, (int)n);
  };
  cast(q,  qb,  NM * D_MODEL);
  cast(k,  kb,  NM * D_MODEL);
  cast(v,  vb,  NM * D_MODEL);
  cast(Wq, wqb, (size_t)D_MODEL * D_MODEL);
  cast(Wk, wkb, (size_t)KV_DIM  * D_MODEL);
  cast(Wv, wvb, (size_t)KV_DIM  * D_MODEL);
  cast(Wo, wob, (size_t)D_MODEL * D_MODEL);

  gemm_bf16<true ><<<dim3(D_MODEL / 64, (unsigned)(NM / 128)), 256, 0, stream>>>(
      qb, wqb, bq, Qp, (int)NM, D_MODEL, D_MODEL);
  gemm_bf16<true ><<<dim3(KV_DIM  / 64, (unsigned)(NM / 128)), 256, 0, stream>>>(
      kb, wkb, bk, Kp, (int)NM, KV_DIM, D_MODEL);
  gemm_bf16<true ><<<dim3(KV_DIM  / 64, (unsigned)(NM / 128)), 256, 0, stream>>>(
      vb, wvb, bv, Vp, (int)NM, KV_DIM, D_MODEL);

  transpose_v<<<(BATCH * NKVH * HDIM * SEQ) / 256, 256, 0, stream>>>(Vp, Vt);

  attn_kernel<<<BATCH * NHEADS * (SEQ / 16), 256, 0, stream>>>(
      Qp, Kp, Vt, out + NM * D_MODEL, ctx);

  gemm_bf16<false><<<dim3(D_MODEL / 64, (unsigned)(NM / 128)), 256, 0, stream>>>(
      ctx, wob, bo, out, (int)NM, D_MODEL, D_MODEL);
}
